// MoEClassificationHead_71545565216953
// MI455X (gfx1250) — matrix operand order repaired and checked by reading, not compiled, not run on hardware
//
#include <hip/hip_runtime.h>

#define B_ 16384
#define D_ 1024
#define H_ 2048
#define E_ 4
#define C_ 3

#define KS_  64   // K elems staged per pipeline stage
#define LDA_ 72   // padded LDS row stride (elems): 144B = 36 banks, conflict-free

typedef __attribute__((ext_vector_type(16))) __bf16 v16bf;
typedef __attribute__((ext_vector_type(8)))  float  v8f;

union FragU { v16bf v; unsigned int u[8]; };

// ---------------------------------------------------------------------------
// async global -> LDS 128-bit copy (CDNA5, tracked by ASYNCcnt)
// ---------------------------------------------------------------------------
__device__ __forceinline__ void async_b128(__bf16* lp, const __bf16* gp) {
    asm volatile("global_load_async_to_lds_b128 %0, %1, off"
                 :
                 : "v"((__attribute__((address_space(3))) __bf16*)lp),
                   "v"((const __attribute__((address_space(1))) __bf16*)gp)
                 : "memory");
}

// ---------------------------------------------------------------------------
// Kernel 1: w1t[e,h,d] = bf16( ex_ln_g[e,d] * w1[e,d,h] )   (tiled transpose)
// grid = E * (D/64) * (H/64), block = 256
// ---------------------------------------------------------------------------
__global__ void k_fold_w1t(const float* __restrict__ w1,
                           const float* __restrict__ exg,
                           __bf16* __restrict__ w1t) {
    const int tilesPerE = (D_ / 64) * (H_ / 64);
    int bid = blockIdx.x;
    int e   = bid / tilesPerE;
    int rem = bid % tilesPerE;
    int dt  = rem / (H_ / 64);
    int ht  = rem % (H_ / 64);
    __shared__ __bf16 tile[64][65];
    int t = threadIdx.x;
#pragma unroll
    for (int i = 0; i < 16; ++i) {
        int idx = i * 256 + t;
        int dl = idx >> 6, hl = idx & 63;
        int dg = dt * 64 + dl, hg = ht * 64 + hl;
        float v = exg[e * D_ + dg] * w1[((size_t)e * D_ + dg) * H_ + hg];
        tile[dl][hl] = (__bf16)v;
    }
    __syncthreads();
#pragma unroll
    for (int i = 0; i < 16; ++i) {
        int idx = i * 256 + t;
        int hl = idx >> 6, dl = idx & 63;
        w1t[((size_t)e * H_ + ht * 64 + hl) * D_ + dt * 64 + dl] = tile[dl][hl];
    }
}

// ---------------------------------------------------------------------------
// Kernel 2: b1f[e,h] = b1[e,h] + sum_d ex_ln_b[e,d] * w1[e,d,h]
// ---------------------------------------------------------------------------
__global__ void k_fold_b1(const float* __restrict__ w1,
                          const float* __restrict__ exlnb,
                          const float* __restrict__ b1,
                          float* __restrict__ b1f) {
    int idx = blockIdx.x * 256 + threadIdx.x;   // 0 .. E*H-1
    int e = idx / H_, h = idx % H_;
    float s = b1[idx];
    for (int d = 0; d < D_; ++d)
        s += exlnb[e * D_ + d] * w1[((size_t)e * D_ + d) * H_ + h];
    b1f[idx] = s;
}

// ---------------------------------------------------------------------------
// Kernel 3: zero partial buffer
// ---------------------------------------------------------------------------
__global__ void k_zero_part(float* __restrict__ part) {
    part[blockIdx.x * 256 + threadIdx.x] = 0.0f;
}

// ---------------------------------------------------------------------------
// Kernel 4: per-row mean/rstd -> normed bf16, gate logits -> softmax weights
// one wave32 per row; block = 256 (8 rows)
// ---------------------------------------------------------------------------
__global__ void k_rows(const float* __restrict__ x,
                       const float* __restrict__ glng,
                       const float* __restrict__ glnb,
                       const float* __restrict__ gate_w,
                       const float* __restrict__ gate_b,
                       __bf16* __restrict__ normed,
                       float* __restrict__ gw) {
    int lane = threadIdx.x & 31;
    int w    = threadIdx.x >> 5;
    int row  = blockIdx.x * 8 + w;
    const float* xr = x + (size_t)row * D_;

    float vals[32];
    float s = 0.f, ss = 0.f;
#pragma unroll
    for (int j = 0; j < 32; ++j) {
        float v = xr[lane + j * 32];
        vals[j] = v;
        s += v; ss += v * v;
    }
#pragma unroll
    for (int m = 1; m <= 16; m <<= 1) {
        s  += __shfl_xor(s,  m, 32);
        ss += __shfl_xor(ss, m, 32);
    }
    float mean = s * (1.0f / D_);
    float var  = ss * (1.0f / D_) - mean * mean;
    float rstd = rsqrtf(var + 1e-5f);

    float gl0 = 0.f, gl1 = 0.f, gl2 = 0.f, gl3 = 0.f;
#pragma unroll
    for (int j = 0; j < 32; ++j) {
        int d = lane + j * 32;
        float nv = (vals[j] - mean) * rstd;
        normed[(size_t)row * D_ + d] = (__bf16)nv;
        float t = nv * glng[d] + glnb[d];
        float4 gv = *(const float4*)(gate_w + (size_t)d * 4);
        gl0 += t * gv.x; gl1 += t * gv.y; gl2 += t * gv.z; gl3 += t * gv.w;
    }
#pragma unroll
    for (int m = 1; m <= 16; m <<= 1) {
        gl0 += __shfl_xor(gl0, m, 32);
        gl1 += __shfl_xor(gl1, m, 32);
        gl2 += __shfl_xor(gl2, m, 32);
        gl3 += __shfl_xor(gl3, m, 32);
    }
    if (lane == 0) {
        float l0 = gl0 + gate_b[0], l1 = gl1 + gate_b[1];
        float l2 = gl2 + gate_b[2], l3 = gl3 + gate_b[3];
        float mx = fmaxf(fmaxf(l0, l1), fmaxf(l2, l3));
        float e0 = expf(l0 - mx), e1 = expf(l1 - mx);
        float e2 = expf(l2 - mx), e3 = expf(l3 - mx);
        float inv = 1.0f / (e0 + e1 + e2 + e3);
        gw[row * 4 + 0] = e0 * inv;
        gw[row * 4 + 1] = e1 * inv;
        gw[row * 4 + 2] = e2 * inv;
        gw[row * 4 + 3] = e3 * inv;
    }
}

__device__ __forceinline__ float gelu_exact(float xv) {
    return 0.5f * xv * (1.0f + erff(xv * 0.70710678118654752f));
}

// ---------------------------------------------------------------------------
// Kernel 5: main fused GEMM, double-buffered async LDS staging, K=64/stage.
// grid = (B/128, E), block = 256 (8 waves: wm=0..3 rows, wn=0..1 cols)
// part[e,b,c] += sum_h gelu(normed@w1t' + b1f) * w2[e,h,c]
// ---------------------------------------------------------------------------
__global__ void k_main(const __bf16* __restrict__ normed,
                       const __bf16* __restrict__ w1t,
                       const float*  __restrict__ b1f,
                       const float*  __restrict__ w2,
                       float* __restrict__ part) {
    __shared__ __bf16 aT[2][128 * LDA_];   // 2 x 18 KB
    __shared__ __bf16 bT[2][128 * LDA_];   // 2 x 18 KB

    const int tid  = threadIdx.x;
    const int lane = tid & 31;
    const int w    = tid >> 5;
    const int wm   = w >> 1;        // 0..3 -> 32-row strip
    const int wn   = w & 1;         // 0..1 -> 64-col strip
    const int row0 = blockIdx.x * 128;
    const int e    = blockIdx.y;

    const int l15 = lane & 15;
    const int hiHalf = (lane >= 16);
    const __bf16* w1tE = w1t + (size_t)e * H_ * D_;

    // lane -> owned row-in-wave permutation (matches where reduced sums land):
    // lanes 0-7 -> rows 0-7, 8-15 -> 16-23, 16-23 -> 8-15, 24-31 -> 24-31
    const int grp = lane >> 3;
    const int rowInWave = lane + ((grp == 1) ? 8 : (grp == 2) ? -8 : 0);
    const int myMt = rowInWave >> 4;
    const int myV  = rowInWave & 7;

    float rowacc[3] = {0.f, 0.f, 0.f};   // only 3 regs live across chunks

    for (int hb = 0; hb < H_; hb += 128) {
        v8f acc[2][4];
#pragma unroll
        for (int mt = 0; mt < 2; ++mt)
#pragma unroll
            for (int nt = 0; nt < 4; ++nt) {
                v8f z = {0.f, 0.f, 0.f, 0.f, 0.f, 0.f, 0.f, 0.f};
                acc[mt][nt] = z;
            }

        // prologue: stage k0 = 0 into buf 0 (8 async b128 per thread)
#pragma unroll
        for (int i = 0; i < 4; ++i) {
            int idx = tid + i * 256;          // 0..1023
            int r   = idx >> 3;               // 0..127
            int seg = (idx & 7) * 8;          // 0..56
            async_b128(&aT[0][r * LDA_ + seg], normed + (size_t)(row0 + r) * D_ + seg);
            async_b128(&bT[0][r * LDA_ + seg], w1tE + (size_t)(hb + r) * D_ + seg);
        }

        int buf = 0;
        for (int s = 0; s < D_ / KS_; ++s) {     // 16 stages of K=64
            if (s < D_ / KS_ - 1) {
                const int kn = (s + 1) * KS_;
                const int nb = buf ^ 1;
#pragma unroll
                for (int i = 0; i < 4; ++i) {
                    int idx = tid + i * 256;
                    int r   = idx >> 3;
                    int seg = (idx & 7) * 8;
                    async_b128(&aT[nb][r * LDA_ + seg], normed + (size_t)(row0 + r) * D_ + kn + seg);
                    async_b128(&bT[nb][r * LDA_ + seg], w1tE + (size_t)(hb + r) * D_ + kn + seg);
                }
                asm volatile("s_wait_asynccnt 0x8" ::: "memory");
            } else {
                asm volatile("s_wait_asynccnt 0x0" ::: "memory");
            }
            __syncthreads();

#pragma unroll
            for (int kk = 0; kk < KS_; kk += 32) {
                // build A fragments (16x32, k = 16*half + s, same order as B)
                v16bf aF[2];
                {
                    const int kg = hiHalf ? 16 : 0;
#pragma unroll
                    for (int mt = 0; mt < 2; ++mt) {
                        int m = wm * 32 + mt * 16 + l15;
                        FragU f;
#pragma unroll
                        for (int v = 0; v < 8; ++v) {
                            int k = kg + 2 * v;
                            f.u[v] = *(const unsigned int*)&aT[buf][m * LDA_ + kk + k];
                        }
                        aF[mt] = f.v;
                    }
                }
                // build B fragments (B 32x16: lanes 0-15 K=0..15, lanes 16-31 K=16..31)
                v16bf bF[4];
                {
                    const int kg = hiHalf ? 16 : 0;
#pragma unroll
                    for (int nt = 0; nt < 4; ++nt) {
                        int n = wn * 64 + nt * 16 + l15;
                        FragU f;
#pragma unroll
                        for (int v = 0; v < 8; ++v)
                            f.u[v] = *(const unsigned int*)&bT[buf][n * LDA_ + kk + kg + 2 * v];
                        bF[nt] = f.v;
                    }
                }
#pragma unroll
                for (int mt = 0; mt < 2; ++mt)
#pragma unroll
                    for (int nt = 0; nt < 4; ++nt)
                        acc[mt][nt] = __builtin_amdgcn_wmma_f32_16x16x32_bf16(
                            false, aF[mt], false, bF[nt],
                            (short)0, acc[mt][nt], false, false);
            }

            __syncthreads();
            buf ^= 1;
        }

        // epilogue (per chunk): bias + exact GELU + contract with w2[h,0:3],
        // then cross-lane reduce and fold into the owned row accumulator.
        float pc[2][8][3] = {};
#pragma unroll
        for (int nt = 0; nt < 4; ++nt) {
            int hg = hb + wn * 64 + nt * 16 + l15;
            float bb  = b1f[e * H_ + hg];
            const float* w2p = w2 + ((size_t)e * H_ + hg) * 3;
            float w20 = w2p[0], w21 = w2p[1], w22 = w2p[2];
#pragma unroll
            for (int mt = 0; mt < 2; ++mt)
#pragma unroll
                for (int v = 0; v < 8; ++v) {
                    float hx = acc[mt][nt][v] + bb;
                    float g  = gelu_exact(hx);
                    pc[mt][v][0] += g * w20;
                    pc[mt][v][1] += g * w21;
                    pc[mt][v][2] += g * w22;
                }
        }
#pragma unroll
        for (int mt = 0; mt < 2; ++mt)
#pragma unroll
            for (int v = 0; v < 8; ++v)
#pragma unroll
                for (int c = 0; c < 3; ++c) {
                    float sv = pc[mt][v][c];
                    sv += __shfl_xor(sv, 1, 32);
                    sv += __shfl_xor(sv, 2, 32);
                    sv += __shfl_xor(sv, 4, 32);
                    sv += __shfl_xor(sv, 8, 32);
                    rowacc[c] += (mt == myMt && v == myV) ? sv : 0.0f;
                }
    }

    // each lane owns one row; exactly 2 atomic contributors per cell (wn=0,1)
    {
        const int grow = row0 + wm * 32 + rowInWave;
#pragma unroll
        for (int c = 0; c < 3; ++c)
            atomicAdd(&part[((size_t)e * B_ + grow) * 3 + c], rowacc[c]);
    }
}

// ---------------------------------------------------------------------------
// Kernel 6: out[b,c] = sum_e gw[b,e] * (part[e,b,c] + b2[e,c])
// ---------------------------------------------------------------------------
__global__ void k_final(const float* __restrict__ part,
                        const float* __restrict__ gw,
                        const float* __restrict__ b2,
                        float* __restrict__ out) {
    int idx = blockIdx.x * 256 + threadIdx.x;   // 0 .. B*3-1
    int b = idx / 3, c = idx % 3;
    float s = 0.f;
#pragma unroll
    for (int e = 0; e < E_; ++e)
        s += gw[b * 4 + e] * (part[((size_t)e * B_ + b) * 3 + c] + b2[e * 3 + c]);
    out[idx] = s;
}

// ---------------------------------------------------------------------------
extern "C" void kernel_launch(void* const* d_in, const int* in_sizes, int n_in,
                              void* d_out, int out_size, void* d_ws, size_t ws_size,
                              hipStream_t stream) {
    const float* x      = (const float*)d_in[0];
    const float* glng   = (const float*)d_in[1];
    const float* glnb   = (const float*)d_in[2];
    const float* gate_w = (const float*)d_in[3];
    const float* gate_b = (const float*)d_in[4];
    const float* exlng  = (const float*)d_in[5];
    const float* exlnb  = (const float*)d_in[6];
    const float* ex_w1  = (const float*)d_in[7];
    const float* ex_b1  = (const float*)d_in[8];
    const float* ex_w2  = (const float*)d_in[9];
    const float* ex_b2  = (const float*)d_in[10];
    float* out = (float*)d_out;

    // workspace carve-up (aligned to 256B)
    char* ws = (char*)d_ws;
    size_t off = 0;
    auto alloc = [&](size_t bytes) {
        void* p = ws + off;
        off = (off + bytes + 255) & ~(size_t)255;
        return p;
    };
    __bf16* normed = (__bf16*)alloc((size_t)B_ * D_ * 2);
    __bf16* w1t    = (__bf16*)alloc((size_t)E_ * H_ * D_ * 2);
    float*  b1f    = (float*) alloc((size_t)E_ * H_ * 4);
    float*  gw     = (float*) alloc((size_t)B_ * E_ * 4);
    float*  part   = (float*) alloc((size_t)E_ * B_ * C_ * 4);
    (void)ws_size; (void)in_sizes; (void)n_in; (void)out_size;

    k_fold_w1t<<<E_ * (D_ / 64) * (H_ / 64), 256, 0, stream>>>(ex_w1, exlng, w1t);
    k_fold_b1 <<<(E_ * H_) / 256, 256, 0, stream>>>(ex_w1, exlnb, ex_b1, b1f);
    k_zero_part<<<(E_ * B_ * C_) / 256, 256, 0, stream>>>(part);
    k_rows<<<B_ / 8, 256, 0, stream>>>(x, glng, glnb, gate_w, gate_b, normed, gw);
    dim3 grid(B_ / 128, E_);
    k_main<<<grid, 256, 0, stream>>>(normed, w1t, b1f, ex_w2, part);
    k_final<<<(B_ * C_) / 256, 256, 0, stream>>>(part, gw, ex_b2, out);
}
